// SelfAttention_47270410060360
// MI455X (gfx1250) — matrix operand order repaired and checked by reading, not compiled, not run on hardware
//
#include <hip/hip_runtime.h>
#include <hip/hip_bf16.h>

typedef _Float16 v16h __attribute__((ext_vector_type(16)));
typedef _Float16 v8h  __attribute__((ext_vector_type(8)));
typedef float    v8f  __attribute__((ext_vector_type(8)));
typedef float    v4f  __attribute__((ext_vector_type(4)));
typedef unsigned int v4u __attribute__((ext_vector_type(4)));
typedef int      v8i  __attribute__((ext_vector_type(8)));
typedef int      v4i  __attribute__((ext_vector_type(4)));

#define ADIM   512
#define AROWS  8192   // 2 * 4096
#define ASEQ   4096
#define AHEADS 8
#define ADHEAD 64

static __device__ __forceinline__ v8f wmma16(v16h a, v16h b, v8f c) {
  // D = A(16x32 f16) * B(32x16 f16) + C(16x16 f32)
  return __builtin_amdgcn_wmma_f32_16x16x32_f16(false, a, false, b, (short)0, c, false, false);
}

static __device__ __forceinline__ v16h mk16(v8h a, v8h b) {
  union { v16h v; v8h h[2]; } u;
  u.h[0] = a; u.h[1] = b;
  return u.v;
}

// Per-lane 16-byte global -> LDS async copy (ASYNCcnt-tracked).
// LDS flat addresses carry the LDS byte offset in the low 32 bits (ISA 10.2),
// so truncating the generic pointer yields the required VDST value.
static __device__ __forceinline__ void async_b128(const _Float16* g, const _Float16* l) {
  asm volatile("global_load_async_to_lds_b128 %0, %1, off"
               :: "v"((unsigned)(size_t)l), "v"((unsigned long long)(size_t)g)
               : "memory");
}

// TDM: DMA a 2D tile (tile_d0 x tile_d1 elements of 2 bytes, row stride
// d0_stride elements) from global into LDS (rows stored consecutively).
// D# packing per CDNA5 ISA 8.3/8.4. TENSORcnt-tracked; EXEC ignored.
static __device__ __forceinline__ void tdm_load_2d(
    const void* gaddr, unsigned lds_off,
    unsigned tile_d0, unsigned tile_d1,
    unsigned tensor_d0, unsigned tensor_d1,
    unsigned long long d0_stride) {
  const unsigned long long ga = (unsigned long long)(size_t)gaddr;
  v4u g0;
  g0[0] = 1u;                                   // count=1, user descriptor
  g0[1] = lds_off;                              // lds_addr (bytes)
  g0[2] = (unsigned)ga;                         // global_addr[31:0]
  g0[3] = (unsigned)(ga >> 32) | (2u << 30);    // global_addr[56:32] | type=2
  v8i g1;
  g1[0] = (int)(1u << 16);                      // wg_mask=0, data_size=1 (2B)
  g1[1] = (int)((tensor_d0 & 0xFFFFu) << 16);   // tensor_dim0[15:0] @ [63:48]
  g1[2] = (int)((tensor_d0 >> 16) | ((tensor_d1 & 0xFFFFu) << 16));
  g1[3] = (int)((tensor_d1 >> 16) | (tile_d0 << 16));   // tile_dim0 @ [127:112]
  g1[4] = (int)(tile_d1 & 0xFFFFu);             // tile_dim1; tile_dim2=0
  g1[5] = (int)(unsigned)(d0_stride & 0xFFFFFFFFull);   // dim0_stride[31:0]
  g1[6] = (int)(unsigned)((d0_stride >> 32) & 0xFFFFull); // [47:32]; dim1_stride=0
  g1[7] = 0;
  const v4i gz = {0, 0, 0, 0};
#if defined(__clang_major__) && (__clang_major__ >= 23)
  const v8i gz8 = {0, 0, 0, 0, 0, 0, 0, 0};
  __builtin_amdgcn_tensor_load_to_lds(g0, g1, gz, gz, gz8, 0);
#else
  __builtin_amdgcn_tensor_load_to_lds(g0, g1, gz, gz, 0);
#endif
}

// ---------------------------------------------------------------------------
// f32 -> f16 pre-conversion (x and the four weight matrices), 8 elems/thread
// ---------------------------------------------------------------------------
__global__ __launch_bounds__(256) void cvt_kernel(const float* __restrict__ src,
                                                  _Float16* __restrict__ dst, int n) {
  const int i = (blockIdx.x * 256 + threadIdx.x) * 8;
  if (i < n) {
    v4f a = *(const v4f*)(src + i);
    v4f b = *(const v4f*)(src + i + 4);
    v8h o;
#pragma unroll
    for (int j = 0; j < 4; ++j) { o[j] = (_Float16)a[j]; o[4 + j] = (_Float16)b[j]; }
    *(v8h*)(dst + i) = o;
  }
}

// ---------------------------------------------------------------------------
// Projection GEMM (all-f16 operands): out = scale * (Xh[8192,512] @ Wh^T)
// transpose_out == 0: out f16 [b,h,n,d]   (Q, K)
// transpose_out == 1: out f16 [b,h,d,n]   (V, pre-transposed for attention)
// One wave computes a 16(M) x 64(N) tile; K-loop step 32.
// A-frag: lane L -> row M=L%16; halves[0..7]=K klo..+7, halves[8..15]=K 16+klo..+7,
//         klo = 8*(L>=16).   B-frag: lane L -> col N=L%16; same K order as A-frag.
// ---------------------------------------------------------------------------
__global__ __launch_bounds__(128) void proj_kernel(
    const _Float16* __restrict__ Xh, const _Float16* __restrict__ Wh,
    _Float16* __restrict__ out, float scale, int transpose_out) {
  const int lane = threadIdx.x & 31;
  const int wave = threadIdx.x >> 5;
  const int wid  = blockIdx.x * 4 + wave;       // 4096 waves total
  const int rb   = (wid >> 3) * 16;             // row-tile base
  const int nb   = (wid & 7) * 64;              // col-tile base
  const int lm   = lane & 15;
  const int hi   = lane >> 4;
  const int klo  = hi * 8;
  const int khi  = hi * 16;

  v8f acc[4] = {};
  const _Float16* arow = Xh + (size_t)(rb + lm) * ADIM;

  for (int kk = 0; kk < ADIM; kk += 32) {
    v16h a = mk16(*(const v8h*)(arow + kk + klo),
                  *(const v8h*)(arow + kk + 16 + klo));
#pragma unroll
    for (int t = 0; t < 4; ++t) {
      const _Float16* wrow = Wh + (size_t)(nb + t * 16 + lm) * ADIM + kk + klo;
      v16h b = mk16(*(const v8h*)wrow, *(const v8h*)(wrow + 16));
      acc[t] = wmma16(a, b, acc[t]);
    }
  }

  // C-frag lane L, VGPR r -> row rb + r + 8*hi, col nb + 16t + lm.
#pragma unroll
  for (int t = 0; t < 4; ++t) {
    const int c = nb + t * 16 + lm;
    const int h = c >> 6;
    const int d = c & 63;
#pragma unroll
    for (int r = 0; r < 8; ++r) {
      const int rr = rb + r + 8 * hi;
      const int b  = rr >> 12;
      const int n  = rr & (ASEQ - 1);
      const _Float16 v = (_Float16)(acc[t][r] * scale);
      if (!transpose_out)
        out[((size_t)(b * AHEADS + h) * ASEQ + n) * ADHEAD + d] = v;
      else
        out[((size_t)(b * AHEADS + h) * ADHEAD + d) * ASEQ + n] = v;
    }
  }
}

// ---------------------------------------------------------------------------
// Flash attention. Block = 4 waves sharing one (b,h); each wave owns 16 query
// rows. Double-buffered staging: K tile via per-lane async-to-LDS copies
// (ASYNCcnt), V^T tile via one TDM 2D descriptor issued by wave 0 (TENSORcnt).
// ---------------------------------------------------------------------------
#define KSTEPS (ASEQ / 32)

static __device__ __forceinline__ void stage_k_async(
    const _Float16* Kb, int kb, int tid, _Float16* ldsK) {
  // K tile: 32 keys x 64 d, fully contiguous in [b,h,n,d]  -> 4 KB
  const _Float16* ks = Kb + (size_t)kb * ADHEAD;
  async_b128(ks + (size_t)tid * 8,         ldsK + tid * 8);
  async_b128(ks + (size_t)(tid + 128) * 8, ldsK + (tid + 128) * 8);
}

__global__ __launch_bounds__(128) void attn_kernel(
    const _Float16* __restrict__ Q, const _Float16* __restrict__ K,
    const _Float16* __restrict__ Vt, _Float16* __restrict__ O) {
  __shared__ __align__(16) _Float16 ldsK[2][32 * 64];    // [buf][key][d]
  __shared__ __align__(16) _Float16 ldsVt[2][64 * 32];   // [buf][d][key]
  __shared__ __align__(16) _Float16 ldsP[4][16 * 32];    // per-wave P staging

  const int tid  = threadIdx.x;
  const int lane = tid & 31;
  const int wave = tid >> 5;
  const int bh   = blockIdx.x >> 6;   // 0..15
  const int qblk = blockIdx.x & 63;   // 0..63
  const int lm   = lane & 15;
  const int hi   = lane >> 4;
  const int qb   = qblk * 64 + wave * 16;

  const _Float16* Qb  = Q  + (size_t)bh * ASEQ * ADHEAD;
  const _Float16* Kb  = K  + (size_t)bh * ASEQ * ADHEAD;
  const _Float16* Vtb = Vt + (size_t)bh * ADHEAD * ASEQ;

  // Q A-fragments for d 0..31 and 32..63 (Q pre-scaled by 1/sqrt(64))
  v16h qa[2];
  {
    const _Float16* qrow = Qb + (size_t)(qb + lm) * ADHEAD;
    const int klo = hi * 8;
#pragma unroll
    for (int f = 0; f < 2; ++f)
      qa[f] = mk16(*(const v8h*)(qrow + f * 32 + klo),
                   *(const v8h*)(qrow + f * 32 + 16 + klo));
  }

  v8f o[4] = {};
  float m[8], l[8];
#pragma unroll
  for (int r = 0; r < 8; ++r) { m[r] = -3.0e38f; l[r] = 0.0f; }

  stage_k_async(Kb, 0, tid, ldsK[0]);
  if (wave == 0)
    tdm_load_2d(Vtb, (unsigned)(size_t)ldsVt[0], 32, ADHEAD, ASEQ, ADHEAD, ASEQ);

  for (int it = 0; it < KSTEPS; ++it) {
    const int cur = it & 1;
    if (wave == 0) __builtin_amdgcn_s_wait_tensorcnt(0);
    asm volatile("s_wait_asynccnt 0x0" ::: "memory");
    __syncthreads();                       // tile `cur` landed; prev compute done
    if (it + 1 < KSTEPS) {
      const int kb2 = (it + 1) * 32;
      stage_k_async(Kb, kb2, tid, ldsK[cur ^ 1]);
      if (wave == 0)
        tdm_load_2d(Vtb + kb2, (unsigned)(size_t)ldsVt[cur ^ 1],
                    32, ADHEAD, ASEQ, ADHEAD, ASEQ);
    }

    const _Float16* tK = ldsK[cur];
    const _Float16* tV = ldsVt[cur];

    // S = Q @ K^T for two 16-key sub-tiles
    v8f s[2] = {};
#pragma unroll
    for (int t = 0; t < 2; ++t) {
#pragma unroll
      for (int f = 0; f < 2; ++f) {
        const _Float16* kp = tK + (t * 16 + lm) * 64 + f * 32 + hi * 8;
        v16h b = mk16(*(const v8h*)kp, *(const v8h*)(kp + 16));
        s[t] = wmma16(qa[f], b, s[t]);
      }
    }

    // Online softmax: row M = r + 8*hi lives in a fixed 16-lane half-group.
    float mnew[8], alpha[8];
#pragma unroll
    for (int r = 0; r < 8; ++r) {
      float v = fmaxf(s[0][r], s[1][r]);
#pragma unroll
      for (int msk = 8; msk >= 1; msk >>= 1)
        v = fmaxf(v, __shfl_xor(v, msk, 32));
      mnew[r]  = fmaxf(m[r], v);
      alpha[r] = __expf(m[r] - mnew[r]);
      m[r]     = mnew[r];
    }
#pragma unroll
    for (int t = 0; t < 2; ++t)
#pragma unroll
      for (int r = 0; r < 8; ++r)
        s[t][r] = __expf(s[t][r] - mnew[r]);
#pragma unroll
    for (int r = 0; r < 8; ++r) {
      float v = s[0][r] + s[1][r];
#pragma unroll
      for (int msk = 8; msk >= 1; msk >>= 1)
        v += __shfl_xor(v, msk, 32);
      l[r] = l[r] * alpha[r] + v;
    }
#pragma unroll
    for (int t = 0; t < 4; ++t)
#pragma unroll
      for (int r = 0; r < 8; ++r)
        o[t][r] *= alpha[r];

    // Re-layout P: C-frag -> row-major LDS -> A-frag (per-wave region)
    _Float16* P = ldsP[wave];
#pragma unroll
    for (int t = 0; t < 2; ++t)
#pragma unroll
      for (int r = 0; r < 8; ++r)
        P[(r + 8 * hi) * 32 + t * 16 + lm] = (_Float16)s[t][r];
    asm volatile("s_wait_dscnt 0x0" ::: "memory");

    v16h pa;
    {
      const _Float16* prow = P + lm * 32 + hi * 8;
      pa = mk16(*(const v8h*)prow, *(const v8h*)(prow + 16));
    }

    // O += P @ V   (4 d-sub-tiles of 16), V^T tile already [d][key]
#pragma unroll
    for (int t = 0; t < 4; ++t) {
      const _Float16* vp = tV + (t * 16 + lm) * 32 + hi * 8;
      v16h b = mk16(*(const v8h*)vp, *(const v8h*)(vp + 16));
      o[t] = wmma16(pa, b, o[t]);
    }
  }

  // Finalize: divide by row sums, store f16 as [b, n, (h d)] = [8192, 512]
  const int b = bh >> 3;
  const int h = bh & 7;
#pragma unroll
  for (int r = 0; r < 8; ++r) {
    const float inv = 1.0f / l[r];
    const int n = qb + r + 8 * hi;
    const size_t rowoff = ((size_t)(b * ASEQ + n)) * ADIM + h * ADHEAD + lm;
#pragma unroll
    for (int t = 0; t < 4; ++t)
      O[rowoff + t * 16] = (_Float16)(o[t][r] * inv);
  }
}

// ---------------------------------------------------------------------------
// Output projection + residual: out_f32[8192,512] = A_f16 @ Wo^T_f16 + X_f32
// ---------------------------------------------------------------------------
__global__ __launch_bounds__(128) void outproj_kernel(
    const _Float16* __restrict__ A, const _Float16* __restrict__ Woh,
    const float* __restrict__ X, float* __restrict__ out) {
  const int lane = threadIdx.x & 31;
  const int wave = threadIdx.x >> 5;
  const int wid  = blockIdx.x * 4 + wave;
  const int rb   = (wid >> 3) * 16;
  const int nb   = (wid & 7) * 64;
  const int lm   = lane & 15;
  const int hi   = lane >> 4;
  const int klo  = hi * 8;
  const int khi  = hi * 16;

  v8f acc[4] = {};
  const _Float16* arow = A + (size_t)(rb + lm) * ADIM;

  for (int kk = 0; kk < ADIM; kk += 32) {
    v16h a = mk16(*(const v8h*)(arow + kk + klo),
                  *(const v8h*)(arow + kk + 16 + klo));
#pragma unroll
    for (int t = 0; t < 4; ++t) {
      const _Float16* wrow = Woh + (size_t)(nb + t * 16 + lm) * ADIM + kk + klo;
      v16h b = mk16(*(const v8h*)wrow, *(const v8h*)(wrow + 16));
      acc[t] = wmma16(a, b, acc[t]);
    }
  }

#pragma unroll
  for (int t = 0; t < 4; ++t) {
    const int c = nb + t * 16 + lm;
#pragma unroll
    for (int r = 0; r < 8; ++r) {
      const size_t rr = rb + r + 8 * hi;
      out[rr * ADIM + c] = acc[t][r] + X[rr * ADIM + c];
    }
  }
}

extern "C" void kernel_launch(void* const* d_in, const int* in_sizes, int n_in,
                              void* d_out, int out_size, void* d_ws, size_t ws_size,
                              hipStream_t stream) {
  const float* x  = (const float*)d_in[0];
  const float* Wq = (const float*)d_in[1];
  const float* Wk = (const float*)d_in[2];
  const float* Wv = (const float*)d_in[3];
  const float* Wo = (const float*)d_in[4];
  float* out = (float*)d_out;

  _Float16* ws = (_Float16*)d_ws;
  const size_t MAT  = (size_t)AROWS * ADIM;  // 8192*512
  const size_t WMAT = (size_t)ADIM * ADIM;   // 512*512
  _Float16* Xh  = ws;
  _Float16* Wqh = ws + MAT;
  _Float16* Wkh = Wqh + WMAT;
  _Float16* Wvh = Wkh + WMAT;
  _Float16* Woh = Wvh + WMAT;
  _Float16* Qh  = Woh + WMAT;
  _Float16* Kh  = Qh + MAT;
  _Float16* Vth = Kh + MAT;
  _Float16* Ah  = Vth + MAT;

  // Pre-convert activations + weights to f16 (removes per-tile v_cvt storms)
  cvt_kernel<<<(int)(MAT / 2048), 256, 0, stream>>>(x,  Xh,  (int)MAT);
  cvt_kernel<<<(int)(WMAT / 2048), 256, 0, stream>>>(Wq, Wqh, (int)WMAT);
  cvt_kernel<<<(int)(WMAT / 2048), 256, 0, stream>>>(Wk, Wkh, (int)WMAT);
  cvt_kernel<<<(int)(WMAT / 2048), 256, 0, stream>>>(Wv, Wvh, (int)WMAT);
  cvt_kernel<<<(int)(WMAT / 2048), 256, 0, stream>>>(Wo, Woh, (int)WMAT);

  dim3 blk(128);
  proj_kernel<<<1024, blk, 0, stream>>>(Xh, Wqh, Qh, 0.125f, 0);  // 1/sqrt(64)
  proj_kernel<<<1024, blk, 0, stream>>>(Xh, Wkh, Kh, 1.0f, 0);
  proj_kernel<<<1024, blk, 0, stream>>>(Xh, Wvh, Vth, 1.0f, 1);   // V transposed
  attn_kernel<<<1024, blk, 0, stream>>>(Qh, Kh, Vth, Ah);
  outproj_kernel<<<1024, blk, 0, stream>>>(Ah, Woh, x, out);
}
